// AggregateModule_60584808677377
// MI455X (gfx1250) — matrix operand order repaired and checked by reading, not compiled, not run on hardware
//
#include <hip/hip_runtime.h>
#include <stdint.h>

// ---------------------------------------------------------------------------
// PointNet++ set-abstraction fused kernel for MI455X (gfx1250, wave32).
//  - ball query: ordered ballot scan, early exit (points are L2-resident)
//  - gather+concat into LDS bf16 tile (32 rows x 160 padded channels)
//  - 3-layer shared MLP via v_wmma_f32_16x16x32_bf16, BN folded to scale/bias
//  - max over K=32 via cross-half shuffle reduction
// ---------------------------------------------------------------------------

#define B_      4
#define N_      20000
#define S_      1024
#define C_      128
#define K_      32
#define RADIUS_ 0.2f
#define EPS_    1e-5f

#define K1P 160   // layer-1 cin padded: 131 -> 160 (5 x K32 WMMA steps)
#define N1  64
#define N2  128
#define N3  256

// workspace layout (bytes); d_ws is hipMalloc'd => >=256B aligned
#define WS_WT1 0                          // bf16 [64][160]  (N-major, padded, row-permuted)
#define WS_WT2 (WS_WT1 + 64*160*2)        // bf16 [128][64]
#define WS_WT3 (WS_WT2 + 128*64*2)        // bf16 [256][128]
#define WS_SC1 (WS_WT3 + 256*128*2)       // f32 [64]
#define WS_BI1 (WS_SC1 + 64*4)
#define WS_SC2 (WS_BI1 + 64*4)
#define WS_BI2 (WS_SC2 + 128*4)
#define WS_SC3 (WS_BI2 + 128*4)
#define WS_BI3 (WS_SC3 + 256*4)

typedef __attribute__((ext_vector_type(16))) __bf16 v16bf;
typedef __attribute__((ext_vector_type(8)))  float  v8f;

__device__ __forceinline__ unsigned short f2bf(float x) {
    union { float f; uint32_t u; } v; v.f = x;
    uint32_t r = v.u + 0x7FFFu + ((v.u >> 16) & 1u);   // round-to-nearest-even
    return (unsigned short)(r >> 16);
}

struct Frag {
    union { v16bf v; uint32_t u[8]; uint4 q[2]; };
};

// A fragment (16x32 bf16, ISA 7.12.2 layout) from row-major LDS tile.
// lane(0..15)=row M, half=lane>>4 selects K groups {half*8..} / {16+half*8..}.
__device__ __forceinline__ void loadA(Frag& f, const unsigned short* lds, int lda,
                                      int rowBase, int kBase, int lane) {
    int m = lane & 15, half = lane >> 4;
    const unsigned short* r = lds + (rowBase + m) * lda + kBase + half * 8;
    const uint32_t* a = reinterpret_cast<const uint32_t*>(r);
    const uint32_t* b = reinterpret_cast<const uint32_t*>(r + 16);
#pragma unroll
    for (int j = 0; j < 4; ++j) { f.u[j] = a[j]; f.u[4 + j] = b[j]; }
}

// B fragment (32x16 bf16): weights stored N-major [n][k] so each lane reads
// K groups {half*8..} / {16+half*8..} like A (two 16-byte global loads, WGP$/L2-hot).
__device__ __forceinline__ void loadB(Frag& f, const unsigned short* wt, int Kpad,
                                      int nBase, int kBase, int lane) {
    int n0 = lane & 15, half = lane >> 4;
    const uint4* p = reinterpret_cast<const uint4*>(wt + (nBase + n0) * Kpad + kBase + half * 8);
    f.q[0] = p[0];
    f.q[1] = p[2];
}

// One MLP layer on the 32-row tile: out = relu(in @ W * scale + bias), bf16 out.
__device__ __forceinline__ void mlp_layer(const unsigned short* inLds, int Kin,
                                          unsigned short* outLds, int Nout,
                                          const unsigned short* wt,
                                          const float* sc, const float* bi, int lane) {
    int n0 = lane & 15, half = lane >> 4;
    for (int mt = 0; mt < 2; ++mt) {
        for (int nt = 0; nt < (Nout >> 4); ++nt) {
            v8f acc = {0.f, 0.f, 0.f, 0.f, 0.f, 0.f, 0.f, 0.f};
            for (int kt = 0; kt < (Kin >> 5); ++kt) {
                Frag a, w;
                loadA(a, inLds, Kin, mt * 16, kt * 32, lane);
                loadB(w, wt, Kin, nt * 16, kt * 32, lane);
                acc = __builtin_amdgcn_wmma_f32_16x16x32_bf16(
                    false, a.v, false, w.v, (short)0, acc, false, false);
            }
            int n = nt * 16 + n0;
            float s = sc[n], o = bi[n];
#pragma unroll
            for (int j = 0; j < 8; ++j) {
                float y = fmaxf(acc[j] * s + o, 0.f);
                outLds[(mt * 16 + half * 8 + j) * Nout + n] = f2bf(y);
            }
        }
    }
}

// ---------------------------------------------------------------------------
// Prep: fold BN into scale/bias; transpose/pad/convert weights to bf16.
// W1 rows permuted: k 0..127 <- features (orig rows 3..130), k 128..130 <- xyz.
// ---------------------------------------------------------------------------
__global__ void prep_kernel(const float* __restrict__ W1, const float* __restrict__ W2,
                            const float* __restrict__ W3,
                            const float* g1, const float* bb1, const float* m1, const float* v1,
                            const float* g2, const float* bb2, const float* m2, const float* v2,
                            const float* g3, const float* bb3, const float* m3, const float* v3,
                            char* __restrict__ ws) {
    int t = blockIdx.x * blockDim.x + threadIdx.x;
    unsigned short* wt1 = (unsigned short*)(ws + WS_WT1);
    unsigned short* wt2 = (unsigned short*)(ws + WS_WT2);
    unsigned short* wt3 = (unsigned short*)(ws + WS_WT3);
    float* sc1 = (float*)(ws + WS_SC1); float* bi1 = (float*)(ws + WS_BI1);
    float* sc2 = (float*)(ws + WS_SC2); float* bi2 = (float*)(ws + WS_BI2);
    float* sc3 = (float*)(ws + WS_SC3); float* bi3 = (float*)(ws + WS_BI3);

    if (t < 64 * K1P) {
        int n = t / K1P, k = t % K1P;
        float val = 0.f;
        if (k < 128)      val = W1[(k + 3) * 64 + n];   // feature channels
        else if (k < 131) val = W1[(k - 128) * 64 + n]; // xyz channels
        wt1[n * K1P + k] = f2bf(val);
    }
    if (t < 128 * 64)  { int n = t / 64,  k = t % 64;  wt2[n * 64  + k] = f2bf(W2[k * 128 + n]); }
    if (t < 256 * 128) { int n = t / 128, k = t % 128; wt3[n * 128 + k] = f2bf(W3[k * 256 + n]); }
    if (t < 448) {
        if (t < 64)       { int c = t;       float s = g1[c] * rsqrtf(v1[c] + EPS_); sc1[c] = s; bi1[c] = bb1[c] - m1[c] * s; }
        else if (t < 192) { int c = t - 64;  float s = g2[c] * rsqrtf(v2[c] + EPS_); sc2[c] = s; bi2[c] = bb2[c] - m2[c] * s; }
        else              { int c = t - 192; float s = g3[c] * rsqrtf(v3[c] + EPS_); sc3[c] = s; bi3[c] = bb3[c] - m3[c] * s; }
    }
}

// ---------------------------------------------------------------------------
// Main fused kernel: 1 wave per ref point (grid = B*S = 4096).
// ---------------------------------------------------------------------------
__global__ void __launch_bounds__(32)
sa_fused_kernel(const float* __restrict__ points, const float* __restrict__ features,
                const float* __restrict__ refs, const char* __restrict__ ws,
                float* __restrict__ out) {
    __shared__ unsigned short sA[K_ * K1P];  // 32x160 bf16 input tile
    __shared__ unsigned short sB[K_ * N1];   // 32x64
    __shared__ unsigned short sC[K_ * N2];   // 32x128
    __shared__ int sIdx[K_];

    const int lane = threadIdx.x;
    const int bs   = blockIdx.x;           // b*S + s
    const int b    = bs >> 10;             // S_ = 1024

    const float* refp = refs + (size_t)bs * 3;
    const float rx = refp[0], ry = refp[1], rz = refp[2];
    const float* pts = points + (size_t)b * N_ * 3;

    // ---- Phase 1: ordered ball query (first K_ hits by index) --------------
    int cnt = 0;
    for (int base = 0; base < N_ && cnt < K_; base += 32) {
        int i = base + lane;
        if (i + 32 < N_) __builtin_prefetch(&pts[(i + 32) * 3], 0, 1);
        bool hit = false;
        if (i < N_) {
            float dx = pts[i * 3 + 0] - rx;
            float dy = pts[i * 3 + 1] - ry;
            float dz = pts[i * 3 + 2] - rz;
            hit = (dx * dx + dy * dy + dz * dz) < (RADIUS_ * RADIUS_);
        }
        unsigned m = (unsigned)__ballot(hit);                 // wave32 mask
        int before = __popc(m & ((1u << lane) - 1u));
        if (hit) { int pos = cnt + before; if (pos < K_) sIdx[pos] = i; }
        cnt += __popc(m);
    }
    __syncthreads();
    int nHit = cnt < K_ ? cnt : K_;
    int firstIdx = (nHit > 0) ? sIdx[0] : (N_ - 1);           // matches jnp clip path
    if (lane >= nHit) sIdx[lane] = firstIdx;
    __syncthreads();

    // ---- Phase 2: gather row `lane` into bf16 LDS tile ---------------------
    {
        int id = sIdx[lane];
        unsigned short* row = &sA[lane * K1P];
        uint32_t* row32 = (uint32_t*)row;
        const float* frow = features + ((size_t)b * N_ + id) * C_;
#pragma unroll 4
        for (int c = 0; c < C_; c += 2) {
            row32[c >> 1] = (uint32_t)f2bf(frow[c]) | ((uint32_t)f2bf(frow[c + 1]) << 16);
        }
        const float invR = 1.0f / RADIUS_;
        float px = pts[id * 3 + 0], py = pts[id * 3 + 1], pz = pts[id * 3 + 2];
        row[128] = f2bf((px - rx) * invR);
        row[129] = f2bf((py - ry) * invR);
        row[130] = f2bf((pz - rz) * invR);
        row[131] = 0;
#pragma unroll
        for (int k = 132; k < K1P; k += 2) row32[k >> 1] = 0;
    }
    __syncthreads();

    // ---- Phase 3: WMMA MLP -------------------------------------------------
    const unsigned short* wt1 = (const unsigned short*)(ws + WS_WT1);
    const unsigned short* wt2 = (const unsigned short*)(ws + WS_WT2);
    const unsigned short* wt3 = (const unsigned short*)(ws + WS_WT3);
    const float* sc1 = (const float*)(ws + WS_SC1); const float* bi1 = (const float*)(ws + WS_BI1);
    const float* sc2 = (const float*)(ws + WS_SC2); const float* bi2 = (const float*)(ws + WS_BI2);
    const float* sc3 = (const float*)(ws + WS_SC3); const float* bi3 = (const float*)(ws + WS_BI3);

    mlp_layer(sA, K1P, sB, N1, wt1, sc1, bi1, lane);   // 131(->160) -> 64
    __syncthreads();
    mlp_layer(sB, N1, sC, N2, wt2, sc2, bi2, lane);    // 64 -> 128
    __syncthreads();

    // layer 3 (128 -> 256) fused with BN + ReLU + max over K_ rows
    float* outp = out + (size_t)bs * N3;
    const int n0 = lane & 15, half = lane >> 4;
    for (int nt = 0; nt < 16; ++nt) {
        int n = nt * 16 + n0;
        float scv = sc3[n], biv = bi3[n];
        float mx = 0.f;                                 // ReLU => max >= 0
        for (int mt = 0; mt < 2; ++mt) {
            v8f acc = {0.f, 0.f, 0.f, 0.f, 0.f, 0.f, 0.f, 0.f};
#pragma unroll
            for (int kt = 0; kt < 4; ++kt) {
                Frag a, w;
                loadA(a, sC, N2, mt * 16, kt * 32, lane);
                loadB(w, wt3, N2, nt * 16, kt * 32, lane);
                acc = __builtin_amdgcn_wmma_f32_16x16x32_bf16(
                    false, a.v, false, w.v, (short)0, acc, false, false);
            }
#pragma unroll
            for (int j = 0; j < 8; ++j) mx = fmaxf(mx, acc[j] * scv + biv);
        }
        mx = fmaxf(mx, __shfl_xor(mx, 16));             // combine M halves
        if (half == 0) outp[n] = mx;
    }
}

// ---------------------------------------------------------------------------
extern "C" void kernel_launch(void* const* d_in, const int* in_sizes, int n_in,
                              void* d_out, int out_size, void* d_ws, size_t ws_size,
                              hipStream_t stream) {
    (void)in_sizes; (void)n_in; (void)out_size; (void)ws_size;
    const float* points   = (const float*)d_in[0];
    const float* features = (const float*)d_in[1];
    const float* refs     = (const float*)d_in[2];
    const float* W1 = (const float*)d_in[3];
    const float* g1 = (const float*)d_in[4];
    const float* b1 = (const float*)d_in[5];
    const float* m1 = (const float*)d_in[6];
    const float* v1 = (const float*)d_in[7];
    const float* W2 = (const float*)d_in[8];
    const float* g2 = (const float*)d_in[9];
    const float* b2 = (const float*)d_in[10];
    const float* m2 = (const float*)d_in[11];
    const float* v2 = (const float*)d_in[12];
    const float* W3 = (const float*)d_in[13];
    const float* g3 = (const float*)d_in[14];
    const float* b3 = (const float*)d_in[15];
    const float* m3 = (const float*)d_in[16];
    const float* v3 = (const float*)d_in[17];

    char* ws = (char*)d_ws;
    prep_kernel<<<128, 256, 0, stream>>>(W1, W2, W3,
                                         g1, b1, m1, v1,
                                         g2, b2, m2, v2,
                                         g3, b3, m3, v3, ws);
    sa_fused_kernel<<<B_ * S_, 32, 0, stream>>>(points, features, refs, ws, (float*)d_out);
}
